// TransformerTemporal_68882685493927
// MI455X (gfx1250) — matrix operand order repaired and checked by reading, not compiled, not run on hardware
//
#include <hip/hip_runtime.h>

// ---------------------------------------------------------------------------
// Types & helpers
// ---------------------------------------------------------------------------
typedef __bf16 bf16;
typedef __attribute__((ext_vector_type(16))) __bf16 v16bf;
typedef __attribute__((ext_vector_type(8)))  float  v8f;
typedef __attribute__((ext_vector_type(4)))  unsigned int u32x4;

union BfPack { u32x4 q[2]; v16bf v; };

__device__ __forceinline__ float bf2f(unsigned short h) {
    unsigned u = ((unsigned)h) << 16; float f; __builtin_memcpy(&f, &u, 4); return f;
}
__device__ __forceinline__ unsigned short f2bf(float f) {
    unsigned u; __builtin_memcpy(&u, &f, 4);
    unsigned r = u + 0x7FFFu + ((u >> 16) & 1u);
    return (unsigned short)(r >> 16);
}

__device__ __forceinline__ v8f wmma_bf16(v16bf a, v16bf b, v8f c) {
    return __builtin_amdgcn_wmma_f32_16x16x32_bf16(false, a, false, b, (short)0, c, false, false);
}

// A-operand (16x32 bf16): lane lm = row M; low lanes K {0..7,16..23}, high lanes K {8..15,24..31}
__device__ __forceinline__ v16bf load_mat_a(const unsigned short* rowptr, int k, int hi) {
    BfPack t;
    const u32x4* p = (const u32x4*)(rowptr + k + (hi ? 8 : 0));
    t.q[0] = p[0];          // 8 elems @ koff
    t.q[1] = p[2];          // 8 elems @ koff+16
    return t.v;
}
// B-operand (32x16 bf16): lane lm = col N; low lanes K {0..7,16..23}, high lanes K {8..15,24..31}
__device__ __forceinline__ v16bf load_mat_b(const unsigned short* rowptr, int k, int hi) {
    BfPack t;
    const u32x4* p = (const u32x4*)(rowptr + k + (hi ? 8 : 0));
    t.q[0] = p[0];
    t.q[1] = p[2];
    return t.v;
}

__device__ __forceinline__ float wave_sum(float v) {
    #pragma unroll
    for (int m = 1; m < 32; m <<= 1) v += __shfl_xor(v, m);
    return v;
}
__device__ __forceinline__ float wave_max(float v) {
    #pragma unroll
    for (int m = 1; m < 32; m <<= 1) v = fmaxf(v, __shfl_xor(v, m));
    return v;
}

// ---------------------------------------------------------------------------
// Generic bf16 WMMA GEMM:  C = act(alpha * A @ B^T + bias)
// A: bf16 [M,K] row-major (lda), B: bf16 [N,K] row-major (ldb) -> used as B^T
// ---------------------------------------------------------------------------
__global__ __launch_bounds__(256) void gemm_bf16_kernel(
    const unsigned short* __restrict__ A, long aBatch, int lda,
    const unsigned short* __restrict__ Bt, long bBatch, int ldb,
    float* __restrict__ Cf, unsigned short* __restrict__ Cb, long cBatch, int ldc,
    const float* __restrict__ bias, int K, float alpha, int relu)
{
    const int lane = threadIdx.x & 31;
    const int w    = threadIdx.x >> 5;
    const int lm   = lane & 15, hi = lane >> 4;
    const int m0   = blockIdx.y * 64  + (w >> 2) * 32;
    const int n0   = blockIdx.x * 128 + (w & 3) * 32;
    const long zb  = blockIdx.z;

    const unsigned short* a0 = A  + zb * aBatch + (long)(m0 + lm) * lda;
    const unsigned short* a1 = a0 + 16L * lda;
    const unsigned short* b0 = Bt + zb * bBatch + (long)(n0 + lm) * ldb;
    const unsigned short* b1 = b0 + 16L * ldb;

    v8f c00 = {}, c01 = {}, c10 = {}, c11 = {};
    for (int k = 0; k < K; k += 32) {
        v16bf av0 = load_mat_a(a0, k, hi);
        v16bf av1 = load_mat_a(a1, k, hi);
        v16bf bv0 = load_mat_b(b0, k, hi);
        v16bf bv1 = load_mat_b(b1, k, hi);
        c00 = wmma_bf16(av0, bv0, c00);
        c01 = wmma_bf16(av0, bv1, c01);
        c10 = wmma_bf16(av1, bv0, c10);
        c11 = wmma_bf16(av1, bv1, c11);
    }

    float bs0 = bias ? bias[n0 + lm]      : 0.f;
    float bs1 = bias ? bias[n0 + 16 + lm] : 0.f;

    auto emit = [&](v8f c, int mb, int ncol, float bv) {
        #pragma unroll
        for (int i = 0; i < 8; i++) {
            float v = alpha * c[i] + bv;
            if (relu) v = fmaxf(v, 0.f);
            long idx = zb * cBatch + (long)(mb + i + 8 * hi) * ldc + ncol;
            if (Cf) Cf[idx] = v;
            if (Cb) Cb[idx] = f2bf(v);
        }
    };
    emit(c00, m0,      n0 + lm,      bs0);
    emit(c01, m0,      n0 + 16 + lm, bs1);
    emit(c10, m0 + 16, n0 + lm,      bs0);
    emit(c11, m0 + 16, n0 + 16 + lm, bs1);
}

// ---------------------------------------------------------------------------
// Flash attention (hd=64, S=1024, rows = token*4 + b).  Q,K: bf16 [4096,512];
// VT: bf16 [b][h][64][1024]; O: bf16 [4096,512].
// ---------------------------------------------------------------------------
__global__ __launch_bounds__(128) void flash_attn_kernel(
    const unsigned short* __restrict__ Q, const unsigned short* __restrict__ Km,
    const unsigned short* __restrict__ VT, unsigned short* __restrict__ O, float scale)
{
    __shared__ unsigned short lds[4][16 * 32];
    const int lane = threadIdx.x & 31, w = threadIdx.x >> 5;
    const int lm = lane & 15, hi = lane >> 4;
    const int b = blockIdx.z, h = blockIdx.y;
    const int q0 = blockIdx.x * 64 + w * 16;

    const unsigned short* qrow = Q + ((long)(q0 + lm) * 4 + b) * 512 + h * 64;
    v16bf aq0 = load_mat_a(qrow, 0, hi);
    v16bf aq1 = load_mat_a(qrow, 32, hi);

    float mr[8], lr[8];
    v8f o0 = {}, o1 = {}, o2 = {}, o3 = {};
    #pragma unroll
    for (int i = 0; i < 8; i++) { mr[i] = -3.0e38f; lr[i] = 0.f; }

    const unsigned short* Kbase = Km + (long)b * 512 + (long)h * 64;
    const unsigned short* Vb = VT + ((long)(b * 8 + h) * 64) * 1024;
    unsigned short* lp = lds[w];

    for (int kt = 0; kt < 1024; kt += 32) {
        const unsigned short* kr0 = Kbase + (long)(kt + lm) * 4 * 512;
        const unsigned short* kr1 = kr0 + 16L * 4 * 512;
        v16bf bk00 = load_mat_b(kr0, 0, hi),  bk01 = load_mat_b(kr0, 32, hi);
        v16bf bk10 = load_mat_b(kr1, 0, hi),  bk11 = load_mat_b(kr1, 32, hi);
        v8f s0 = {}, s1 = {};
        s0 = wmma_bf16(aq0, bk00, s0); s0 = wmma_bf16(aq1, bk01, s0);
        s1 = wmma_bf16(aq0, bk10, s1); s1 = wmma_bf16(aq1, bk11, s1);

        float p0[8], p1[8];
        #pragma unroll
        for (int i = 0; i < 8; i++) {
            float x0 = s0[i] * scale, x1 = s1[i] * scale;
            float t = fmaxf(x0, x1);
            t = fmaxf(t, __shfl_xor(t, 1));
            t = fmaxf(t, __shfl_xor(t, 2));
            t = fmaxf(t, __shfl_xor(t, 4));
            t = fmaxf(t, __shfl_xor(t, 8));
            float nm = fmaxf(mr[i], t);
            float al = __expf(mr[i] - nm);
            float e0 = __expf(x0 - nm), e1 = __expf(x1 - nm);
            float rs = e0 + e1;
            rs += __shfl_xor(rs, 1); rs += __shfl_xor(rs, 2);
            rs += __shfl_xor(rs, 4); rs += __shfl_xor(rs, 8);
            lr[i] = lr[i] * al + rs; mr[i] = nm;
            o0[i] *= al; o1[i] *= al; o2[i] *= al; o3[i] *= al;
            p0[i] = e0; p1[i] = e1;
        }
        // stage P tile [16 q x 32 k] into LDS, reload in A layout
        #pragma unroll
        for (int i = 0; i < 8; i++) {
            int r = i + 8 * hi;
            lp[r * 32 + lm]      = f2bf(p0[i]);
            lp[r * 32 + 16 + lm] = f2bf(p1[i]);
        }
        asm volatile("s_wait_dscnt 0" ::: "memory");
        BfPack t;
        const u32x4* lq = (const u32x4*)(lp + lm * 32 + (hi ? 8 : 0));
        t.q[0] = lq[0]; t.q[1] = lq[2];
        v16bf ap = t.v;

        v16bf bv0 = load_mat_b(Vb + (long)(0  + lm) * 1024, kt, hi);
        v16bf bv1 = load_mat_b(Vb + (long)(16 + lm) * 1024, kt, hi);
        v16bf bv2 = load_mat_b(Vb + (long)(32 + lm) * 1024, kt, hi);
        v16bf bv3 = load_mat_b(Vb + (long)(48 + lm) * 1024, kt, hi);
        o0 = wmma_bf16(ap, bv0, o0);
        o1 = wmma_bf16(ap, bv1, o1);
        o2 = wmma_bf16(ap, bv2, o2);
        o3 = wmma_bf16(ap, bv3, o3);
    }

    #pragma unroll
    for (int i = 0; i < 8; i++) {
        float inv = 1.f / lr[i];
        int row = q0 + i + 8 * hi;
        long base = ((long)row * 4 + b) * 512 + h * 64;
        O[base + lm]        = f2bf(o0[i] * inv);
        O[base + 16 + lm]   = f2bf(o1[i] * inv);
        O[base + 32 + lm]   = f2bf(o2[i] * inv);
        O[base + 48 + lm]   = f2bf(o3[i] * inv);
    }
}

// ---------------------------------------------------------------------------
// LayerNorm over D=512; one wave per row (8 rows / block).
// x = A            (srow==0, Badd==0)
// x = A + Badd     (Badd!=0)
// x = A * srow[row](srow!=0)
// ---------------------------------------------------------------------------
__global__ __launch_bounds__(256) void ln_kernel(
    const float* __restrict__ A, const float* __restrict__ Badd,
    const float* __restrict__ srow, const float* __restrict__ w,
    const float* __restrict__ bias, float* __restrict__ outF,
    unsigned short* __restrict__ outB)
{
    const int lane = threadIdx.x & 31, wv = threadIdx.x >> 5;
    const long row = (long)blockIdx.x * 8 + wv;
    const float* a = A + row * 512;
    float s = srow ? srow[row] : 0.f;
    float x[16];
    float acc = 0.f;
    #pragma unroll
    for (int j = 0; j < 16; j++) {
        int c = lane + 32 * j;
        float v = a[c];
        if (srow) v *= s;
        else if (Badd) v += Badd[row * 512 + c];
        x[j] = v; acc += v;
    }
    float mean = wave_sum(acc) * (1.f / 512.f);
    float v2 = 0.f;
    #pragma unroll
    for (int j = 0; j < 16; j++) { float d = x[j] - mean; v2 += d * d; }
    float var = wave_sum(v2) * (1.f / 512.f);
    float rs = rsqrtf(var + 1e-5f);
    #pragma unroll
    for (int j = 0; j < 16; j++) {
        int c = lane + 32 * j;
        float y = (x[j] - mean) * rs * w[c] + bias[c];
        if (outF) outF[row * 512 + c] = y;
        if (outB) outB[row * 512 + c] = f2bf(y);
    }
}

// L2-normalize rows of [4096, 2048] f32 -> bf16
__global__ __launch_bounds__(256) void l2norm_kernel(
    const float* __restrict__ X, unsigned short* __restrict__ Y)
{
    const int lane = threadIdx.x & 31, wv = threadIdx.x >> 5;
    const long row = (long)blockIdx.x * 8 + wv;
    const float* x = X + row * 2048;
    float ss = 0.f;
    for (int j = 0; j < 64; j++) { float v = x[lane + 32 * j]; ss += v * v; }
    ss = wave_sum(ss);
    float inv = 1.f / fmaxf(sqrtf(ss), 1e-12f);
    for (int j = 0; j < 64; j++) {
        int c = lane + 32 * j;
        Y[row * 2048 + c] = f2bf(x[c] * inv);
    }
}

// Fusion softmax over s (len 1024); rows = b*1024 + l. Also O1[l*4+b] = sum p*rt.
__global__ __launch_bounds__(256) void fus_softmax_kernel(
    const float* __restrict__ S, unsigned short* __restrict__ P,
    const float* __restrict__ RT, float* __restrict__ O1)
{
    const int lane = threadIdx.x & 31, wv = threadIdx.x >> 5;
    const int row = blockIdx.x * 8 + wv;
    const int b = row >> 10, l = row & 1023;
    const float* s = S + (long)row * 1024;
    float x[32];
    float mx = -3.0e38f;
    #pragma unroll
    for (int j = 0; j < 32; j++) { x[j] = s[lane + 32 * j]; mx = fmaxf(mx, x[j]); }
    mx = wave_max(mx);
    float sm = 0.f, srt = 0.f;
    #pragma unroll
    for (int j = 0; j < 32; j++) {
        float e = __expf(x[j] - mx);
        x[j] = e; sm += e;
        srt += e * RT[(lane + 32 * j) * 4 + b];
    }
    sm = wave_sum(sm); srt = wave_sum(srt);
    float inv = 1.f / sm;
    #pragma unroll
    for (int j = 0; j < 32; j++)
        P[(long)row * 1024 + lane + 32 * j] = f2bf(x[j] * inv);
    if (lane == 0) O1[l * 4 + b] = srt * inv;
}

// ---------------------------------------------------------------------------
// Packing / elementwise kernels
// ---------------------------------------------------------------------------
__global__ void pack_temp_kernel(const float* __restrict__ src, const float* __restrict__ pos,
                                 const float* __restrict__ resp,
                                 float* __restrict__ ST, float* __restrict__ PT,
                                 float* __restrict__ RT)
{
    long idx = (long)blockIdx.x * 256 + threadIdx.x;
    if (idx < 2097152) {
        int b = idx >> 19, t = (idx >> 17) & 3, c = (idx >> 8) & 511, s = idx & 255;
        long row = (long)(t * 256 + s) * 4 + b;
        ST[row * 512 + c] = src[idx];
        PT[row * 512 + c] = pos[idx];
    }
    if (idx < 4096) {
        int b = idx >> 10, t = (idx >> 8) & 3, s = idx & 255;
        RT[(t * 256 + s) * 4 + b] = resp[idx];
    }
}

__global__ void pack_search_kernel(const float* __restrict__ src, const float* __restrict__ pos,
                                   float* __restrict__ SS, float* __restrict__ PS)
{
    long idx = (long)blockIdx.x * 256 + threadIdx.x;
    if (idx >= 2097152) return;
    int b = idx >> 19, c = (idx >> 10) & 511, s = idx & 1023;
    long row = (long)s * 4 + b;
    SS[row * 512 + c] = src[idx];
    PS[row * 512 + c] = pos[idx];
}

__global__ void cvt_bf16_kernel(const float* __restrict__ in, unsigned short* __restrict__ out, long n)
{
    long idx = (long)blockIdx.x * 256 + threadIdx.x;
    if (idx < n) out[idx] = f2bf(in[idx]);
}

__global__ void add2bf16_kernel(const float* __restrict__ a, const float* __restrict__ b,
                                unsigned short* __restrict__ out, long n)
{
    long idx = (long)blockIdx.x * 256 + threadIdx.x;
    if (idx < n) out[idx] = f2bf(a[idx] + b[idx]);
}

// V [4096,512] bf16 -> VT [b][h][64][1024] bf16
__global__ void vtrans_kernel(const unsigned short* __restrict__ V, unsigned short* __restrict__ VT)
{
    long idx = (long)blockIdx.x * 256 + threadIdx.x;
    if (idx >= 2097152) return;
    int s = idx & 1023, d = (idx >> 10) & 63, h = (idx >> 16) & 7, b = (idx >> 19) & 3;
    VT[idx] = V[((long)(s * 4 + b)) * 512 + h * 64 + d];
}

// Build key = st*(1+rt) (f32 + bf16), st bf16, and kv^T[b][e][s] = bf16(key*rt)
__global__ void fus_prep_kernel(const float* __restrict__ ST, const float* __restrict__ RT,
                                float* __restrict__ KEYF, unsigned short* __restrict__ KEYB,
                                unsigned short* __restrict__ STB, unsigned short* __restrict__ KVT)
{
    long idx = (long)blockIdx.x * 256 + threadIdx.x;
    if (idx >= 2097152) return;
    long row = idx >> 9; int c = idx & 511;
    int b = row & 3; int l = (int)(row >> 2);
    float st = ST[idx]; float r = RT[row];
    float key = st * (1.f + r);
    KEYF[idx] = key;
    KEYB[idx] = f2bf(key);
    STB[idx]  = f2bf(st);
    KVT[((long)(b * 512 + c) << 10) + l] = f2bf(key * r);
}

// Final head: out[b*1024+s] = dot(X[row,0:256], w3) + b3,  row = s*4+b
__global__ __launch_bounds__(256) void head_out_kernel(
    const unsigned short* __restrict__ X, const float* __restrict__ w3,
    const float* __restrict__ b3, float* __restrict__ out)
{
    const int lane = threadIdx.x & 31, wv = threadIdx.x >> 5;
    const int row = blockIdx.x * 8 + wv;
    const unsigned short* x = X + (long)row * 256;
    float acc = 0.f;
    #pragma unroll
    for (int j = 0; j < 8; j++) { int c = lane + 32 * j; acc += bf2f(x[c]) * w3[c]; }
    acc = wave_sum(acc);
    if (lane == 0) {
        int b = row & 3, s = row >> 2;
        out[b * 1024 + s] = acc + b3[0];
    }
}

// ---------------------------------------------------------------------------
// Host orchestration
// ---------------------------------------------------------------------------
extern "C" void kernel_launch(void* const* d_in, const int* in_sizes, int n_in,
                              void* d_out, int out_size, void* d_ws, size_t ws_size,
                              hipStream_t stream)
{
    (void)in_sizes; (void)n_in; (void)out_size; (void)ws_size;
    constexpr long MD  = 4096L * 512;
    constexpr long MKD = 4096L * 2048;

    // ---- inputs / params ----
    const float* src_temp   = (const float*)d_in[0];
    const float* pos_temp   = (const float*)d_in[2];
    const float* resp       = (const float*)d_in[3];
    const float* src_search = (const float*)d_in[4];
    const float* pos_search = (const float*)d_in[6];
    auto P = [&](int i) { return (const float*)d_in[i]; };
    // enc: 7..18, dec: 19..30
    const float *dnw = P(31), *dnb = P(32);
    const float *WKw = P(33), *WKb = P(34), *WQw = P(35), *WQb = P(36);
    const float *fn1w = P(37), *fn1b = P(38), *fn2w = P(39), *fn2b = P(40), *fn3w = P(41), *fn3b = P(42);
    const float *flinw = P(43), *flinb = P(44);
    const float *m1w = P(45), *m1b = P(46), *m2w = P(47), *m2b = P(48), *m3w = P(49), *m3b = P(50);

    // ---- workspace carve ----
    char* ws = (char*)d_ws;
    size_t off = 0;
    auto alloc = [&](size_t n) { size_t o = off; off = (off + n + 255) & ~(size_t)255; return o; };
    float* ST  = (float*)(ws + alloc(MD * 4));
    float* PT  = (float*)(ws + alloc(MD * 4));
    float* SS  = (float*)(ws + alloc(MD * 4));
    float* PS  = (float*)(ws + alloc(MD * 4));
    float* RT  = (float*)(ws + alloc(4096 * 4));
    float* XA  = (float*)(ws + alloc(MD * 4));
    float* XB  = (float*)(ws + alloc(MD * 4));
    float* XC  = (float*)(ws + alloc(MD * 4));
    float* O1  = (float*)(ws + alloc(4096 * 4));
    float* WQF = (float*)(ws + alloc(MKD * 4));   // reused as AFF (16MB <= 32MB)
    float* WKF = (float*)(ws + alloc(MKD * 4));   // reused as AFFB (8MB <= 32MB)
    unsigned short* BFX  = (unsigned short*)(ws + alloc(MD * 2));
    unsigned short* BFY  = (unsigned short*)(ws + alloc(MD * 2));
    unsigned short* BFQ  = (unsigned short*)(ws + alloc(MD * 2));
    unsigned short* BFK  = (unsigned short*)(ws + alloc(MD * 2));
    unsigned short* BFV  = (unsigned short*)(ws + alloc(MD * 2));
    unsigned short* VB   = (unsigned short*)(ws + alloc(MD * 2));
    unsigned short* OATT = (unsigned short*)(ws + alloc(MD * 2));
    unsigned short* VT   = (unsigned short*)(ws + alloc(MD * 2));
    unsigned short* HBF  = (unsigned short*)(ws + alloc(MKD * 2));
    unsigned short* WQB  = (unsigned short*)(ws + alloc(MKD * 2));
    unsigned short* WKB  = (unsigned short*)(ws + alloc(MKD * 2));
    unsigned short* KVT  = (unsigned short*)(ws + alloc(4L * 512 * 1024 * 2));
    unsigned short* MH1  = (unsigned short*)(ws + alloc(4096L * 256 * 2));
    unsigned short* MH2  = (unsigned short*)(ws + alloc(4096L * 256 * 2));
    unsigned short* WB_ENC_IN  = (unsigned short*)(ws + alloc(1536L * 512 * 2));
    unsigned short* WB_ENC_OUT = (unsigned short*)(ws + alloc(512L * 512 * 2));
    unsigned short* WB_ENC_L1  = (unsigned short*)(ws + alloc(2048L * 512 * 2));
    unsigned short* WB_ENC_L2  = (unsigned short*)(ws + alloc(512L * 2048 * 2));
    unsigned short* WB_DEC_IN  = (unsigned short*)(ws + alloc(1536L * 512 * 2));
    unsigned short* WB_DEC_OUT = (unsigned short*)(ws + alloc(512L * 512 * 2));
    unsigned short* WB_DEC_L1  = (unsigned short*)(ws + alloc(2048L * 512 * 2));
    unsigned short* WB_DEC_L2  = (unsigned short*)(ws + alloc(512L * 2048 * 2));
    unsigned short* WB_WK   = (unsigned short*)(ws + alloc(2048L * 512 * 2));
    unsigned short* WB_WQ   = (unsigned short*)(ws + alloc(2048L * 512 * 2));
    unsigned short* WB_FLIN = (unsigned short*)(ws + alloc(512L * 512 * 2));
    unsigned short* WB_M1   = (unsigned short*)(ws + alloc(256L * 512 * 2));
    unsigned short* WB_M2   = (unsigned short*)(ws + alloc(256L * 256 * 2));
    float*          AFF  = WQF;                       // [4][1024][1024] f32
    unsigned short* AFFB = (unsigned short*)WKF;      // [4][1024][1024] bf16

    auto cvtw = [&](const float* in, unsigned short* outp, long n) {
        cvt_bf16_kernel<<<(unsigned)((n + 255) / 256), 256, 0, stream>>>(in, outp, n);
    };
    auto gemm = [&](const unsigned short* A, long aB, int lda,
                    const unsigned short* Bt, long bB, int ldb,
                    float* Cf, unsigned short* Cb, long cB, int ldc,
                    const float* bias, int M, int N, int K, float alpha, int relu, int batches) {
        gemm_bf16_kernel<<<dim3(N / 128, M / 64, batches), 256, 0, stream>>>(
            A, aB, lda, Bt, bB, ldb, Cf, Cb, cB, ldc, bias, K, alpha, relu);
    };

    // ---- convert weights to bf16 ----
    cvtw(P(7),  WB_ENC_IN,  1536L * 512);
    cvtw(P(9),  WB_ENC_OUT, 512L * 512);
    cvtw(P(11), WB_ENC_L1,  2048L * 512);
    cvtw(P(13), WB_ENC_L2,  512L * 2048);
    cvtw(P(19), WB_DEC_IN,  1536L * 512);
    cvtw(P(21), WB_DEC_OUT, 512L * 512);
    cvtw(P(23), WB_DEC_L1,  2048L * 512);
    cvtw(P(25), WB_DEC_L2,  512L * 2048);
    cvtw(WKw,   WB_WK,      2048L * 512);
    cvtw(WQw,   WB_WQ,      2048L * 512);
    cvtw(flinw, WB_FLIN,    512L * 512);
    cvtw(m1w,   WB_M1,      256L * 512);
    cvtw(m2w,   WB_M2,      256L * 256);

    // ---- pack tokens ----
    pack_temp_kernel<<<8192, 256, 0, stream>>>(src_temp, pos_temp, resp, ST, PT, RT);
    pack_search_kernel<<<8192, 256, 0, stream>>>(src_search, pos_search, SS, PS);

    // ---- shared attention layer runner ----
    auto run_attn = [&](const float* qsrc, const float* posq, const float* ksrc, const float* posk,
                        const float* vsrc,
                        const unsigned short* win, const float* winb,
                        const unsigned short* wout, const float* woutb,
                        const unsigned short* wl1, const float* l1b,
                        const unsigned short* wl2, const float* l2b,
                        const float* n1w, const float* n1b, const float* n2w, const float* n2b,
                        float* dstF, unsigned short* dstB) {
        add2bf16_kernel<<<8192, 256, 0, stream>>>(qsrc, posq, BFX, MD);
        const unsigned short* kA = BFX;
        if (ksrc != qsrc || posk != posq) {
            add2bf16_kernel<<<8192, 256, 0, stream>>>(ksrc, posk, BFY, MD);
            kA = BFY;
        }
        cvt_bf16_kernel<<<8192, 256, 0, stream>>>(vsrc, VB, MD);
        gemm(BFX, 0, 512, win,              0, 512, nullptr, BFQ, 0, 512, winb,        4096, 512, 512, 1.f, 0, 1);
        gemm(kA,  0, 512, win + 512L * 512, 0, 512, nullptr, BFK, 0, 512, winb + 512,  4096, 512, 512, 1.f, 0, 1);
        gemm(VB,  0, 512, win + 1024L * 512,0, 512, nullptr, BFV, 0, 512, winb + 1024, 4096, 512, 512, 1.f, 0, 1);
        vtrans_kernel<<<8192, 256, 0, stream>>>(BFV, VT);
        flash_attn_kernel<<<dim3(16, 8, 4), 128, 0, stream>>>(BFQ, BFK, VT, OATT, 0.125f);
        gemm(OATT, 0, 512, wout, 0, 512, XA, nullptr, 0, 512, woutb, 4096, 512, 512, 1.f, 0, 1);
        ln_kernel<<<512, 256, 0, stream>>>(qsrc, XA, nullptr, n1w, n1b, XB, BFY);
        gemm(BFY, 0, 512,  wl1, 0, 512,  nullptr, HBF, 0, 2048, l1b, 4096, 2048, 512, 1.f, 1, 1);
        gemm(HBF, 0, 2048, wl2, 0, 2048, XA, nullptr, 0, 512,  l2b, 4096, 512, 2048, 1.f, 0, 1);
        ln_kernel<<<512, 256, 0, stream>>>(XB, XA, nullptr, n2w, n2b, dstF, dstB);
    };

    // ---- encoder (shared weights) on template and search ----
    run_attn(ST, PT, ST, PT, ST, WB_ENC_IN, P(8), WB_ENC_OUT, P(10), WB_ENC_L1, P(12),
             WB_ENC_L2, P(14), P(15), P(16), P(17), P(18), ST, nullptr);
    run_attn(SS, PS, SS, PS, SS, WB_ENC_IN, P(8), WB_ENC_OUT, P(10), WB_ENC_L1, P(12),
             WB_ENC_L2, P(14), P(15), P(16), P(17), P(18), SS, nullptr);

    // ---- fusion ----
    fus_prep_kernel<<<8192, 256, 0, stream>>>(ST, RT, XC, BFY, BFX, KVT);
    gemm(BFX, 0, 512, WB_WQ, 0, 512, WQF, nullptr, 0, 2048, WQb, 4096, 2048, 512, 1.f, 0, 1);
    gemm(BFY, 0, 512, WB_WK, 0, 512, WKF, nullptr, 0, 2048, WKb, 4096, 2048, 512, 1.f, 0, 1);
    l2norm_kernel<<<512, 256, 0, stream>>>(WQF, WQB);
    l2norm_kernel<<<512, 256, 0, stream>>>(WKF, WKB);
    // scores: per-batch [1024,2048] x [1024,2048]^T * 30   (WQF/WKF now reusable)
    gemm(WQB, 2048, 8192, WKB, 2048, 8192, AFF, nullptr, 1048576, 1024, nullptr,
         1024, 1024, 2048, 30.f, 0, 4);
    fus_softmax_kernel<<<512, 256, 0, stream>>>(AFF, AFFB, RT, O1);
    // PV: aff @ kv  ->  o2 rows (l*4+b)
    gemm(AFFB, 1048576, 1024, KVT, 524288, 1024, XA, nullptr, 512, 2048, nullptr,
         1024, 512, 1024, 1.f, 0, 4);
    ln_kernel<<<512, 256, 0, stream>>>(ST, nullptr, O1, fn1w, fn1b, XB, nullptr);  // f1
    ln_kernel<<<512, 256, 0, stream>>>(ST, XA, nullptr, fn2w, fn2b, XC, nullptr);  // f2
    add2bf16_kernel<<<8192, 256, 0, stream>>>(XB, XC, BFX, MD);
    gemm(BFX, 0, 512, WB_FLIN, 0, 512, XA, nullptr, 0, 512, flinb, 4096, 512, 512, 1.f, 0, 1);
    ln_kernel<<<512, 256, 0, stream>>>(XA, nullptr, nullptr, fn3w, fn3b, ST, nullptr);

    // ---- decoder (cross-attention) ----
    run_attn(SS, PS, ST, PT, ST, WB_DEC_IN, P(20), WB_DEC_OUT, P(22), WB_DEC_L1, P(24),
             WB_DEC_L2, P(26), P(27), P(28), P(29), P(30), SS, nullptr);
    ln_kernel<<<512, 256, 0, stream>>>(SS, nullptr, nullptr, dnw, dnb, nullptr, BFX);

    // ---- MLP head ----
    gemm(BFX, 0, 512, WB_M1, 0, 512, nullptr, MH1, 0, 256, m1b, 4096, 256, 512, 1.f, 1, 1);
    gemm(MH1, 0, 256, WB_M2, 0, 256, nullptr, MH2, 0, 256, m2b, 4096, 256, 256, 1.f, 1, 1);
    head_out_kernel<<<512, 256, 0, stream>>>(MH2, m3w, m3b, (float*)d_out);
}
